// TransformerEncoder1D_1331439861877
// MI455X (gfx1250) — matrix operand order repaired and checked by reading, not compiled, not run on hardware
//
#include <hip/hip_runtime.h>

#define Bn    32
#define SEQ   744
#define DM    512
#define NH    8
#define HDm   64
#define NL    4
#define EXPF  4
#define MROWS (Bn*SEQ)   // 23808 = 186*128

typedef __attribute__((ext_vector_type(16))) __bf16          v16bf;
typedef __attribute__((ext_vector_type(8)))  float           v8f;
typedef __attribute__((ext_vector_type(8)))  unsigned short  v8us;
typedef __attribute__((ext_vector_type(16))) unsigned short  v16us;

union BFV { v16us u; v16bf b; v8us h[2]; };

__device__ __forceinline__ unsigned short f2bf(float f){
  unsigned int v = __builtin_bit_cast(unsigned int, f);
  v += 0x7FFFu + ((v >> 16) & 1u);          // round-to-nearest-even
  return (unsigned short)(v >> 16);
}

__device__ __forceinline__ v8f wmma_bf16(v16bf a, v16bf b, v8f c){
  // D = A(16x32 bf16) x B(32x16 bf16) + C(16x16 f32)
  return __builtin_amdgcn_wmma_f32_16x16x32_bf16(false, a, false, b,
                                                 (short)0, c, false, false);
}

// async global->LDS copy of 16 bytes per lane (gfx1250, tracked by ASYNCcnt)
__device__ __forceinline__ void async_copy_b128(const void* gsrc, void* ldst){
  const unsigned long long ga = (unsigned long long)(uintptr_t)gsrc;
  const unsigned la = (unsigned)(uintptr_t)ldst;   // low 32 bits = LDS offset
  asm volatile("global_load_async_to_lds_b128 %0, %1, off"
               :: "v"(la), "v"(ga) : "memory");
}

// ---------------------------------------------------------------------------
// f32 -> bf16 bulk convert (weights)
// ---------------------------------------------------------------------------
__global__ __launch_bounds__(256)
void cvt_kernel(const float* __restrict__ in, unsigned short* __restrict__ out, long n){
  long i = (long)blockIdx.x * 256 + threadIdx.x;
  if (i < n) out[i] = f2bf(in[i]);
}

// ---------------------------------------------------------------------------
// embedding + positional encoding:  h = emb[x]*sqrt(D) + pe   (f32 + bf16)
// ---------------------------------------------------------------------------
__global__ __launch_bounds__(256)
void embed_kernel(const int* __restrict__ x, const float* __restrict__ emb,
                  const float* __restrict__ pe, float* __restrict__ hf,
                  unsigned short* __restrict__ hb){
  const int row = blockIdx.x;              // 0..B*S-1
  const int s   = row % SEQ;
  const int id  = x[row];
  const size_t base = (size_t)row * DM;
  for (int i = threadIdx.x; i < DM; i += 256){
    float v = emb[(size_t)id*DM + i] * 22.627416997969522f + pe[(size_t)s*DM + i];
    hf[base + i] = v;
    hb[base + i] = f2bf(v);
  }
}

// ---------------------------------------------------------------------------
// one 16x64 projection tile: out = A(16x64) x W(64x64)^T
// transposed==false: out[s*64 + col]   (Q, K:  [S,64] per head)
// transposed==true : out[col*SEQ + s]  (Vt:    [64,S] per head)
// ---------------------------------------------------------------------------
__device__ __forceinline__ void qkv_one(const v16bf* af,
                                        const unsigned short* __restrict__ W,
                                        unsigned short* __restrict__ ob,
                                        int m0, int lane, bool transposed){
  v8f acc[4];
  #pragma unroll
  for (int j = 0; j < 4; ++j) acc[j] = v8f{};
  #pragma unroll
  for (int kk = 0; kk < 2; ++kk){
    #pragma unroll
    for (int j = 0; j < 4; ++j){
      BFV r;  // B = W^T, W row-major [64,64] -> two 8-bf16 chunks per lane (k 8h.., 16+8h..)
      const unsigned short* p = W + (size_t)(j*16 + (lane & 15))*HDm + kk*32 + ((lane & 16) >> 1);
      r.h[0] = *(const v8us*)(p);
      r.h[1] = *(const v8us*)(p + 16);
      acc[j] = wmma_bf16(af[kk], r.b, acc[j]);
    }
  }
  const int rb = (lane & 16) ? 8 : 0;
  const int cl = lane & 15;
  if (m0 + 16 <= SEQ){                      // uniform fast path
    #pragma unroll
    for (int e = 0; e < 8; ++e){
      const int s = m0 + rb + e;
      #pragma unroll
      for (int j = 0; j < 4; ++j){
        const int col = j*16 + cl;
        const size_t idx = transposed ? ((size_t)col*SEQ + s) : ((size_t)s*HDm + col);
        ob[idx] = f2bf(acc[j][e]);
      }
    }
  } else {
    #pragma unroll
    for (int e = 0; e < 8; ++e){
      const int s = m0 + rb + e;
      if (s < SEQ){
        #pragma unroll
        for (int j = 0; j < 4; ++j){
          const int col = j*16 + cl;
          const size_t idx = transposed ? ((size_t)col*SEQ + s) : ((size_t)s*HDm + col);
          ob[idx] = f2bf(acc[j][e]);
        }
      }
    }
  }
}

// ---------------------------------------------------------------------------
// per-head QKV projection; V is written transposed [B,H,64,S]
// ---------------------------------------------------------------------------
__global__ __launch_bounds__(256)
void qkv_kernel(const unsigned short* __restrict__ hbf,
                const unsigned short* __restrict__ wq,
                const unsigned short* __restrict__ wk,
                const unsigned short* __restrict__ wv,
                unsigned short* __restrict__ qo,
                unsigned short* __restrict__ ko,
                unsigned short* __restrict__ vo){
  const int lane = threadIdx.x & 31;
  const int wave = threadIdx.x >> 5;
  const int bh   = blockIdx.y;
  const int b    = bh >> 3, h = bh & 7;
  const int m0   = blockIdx.x * 128 + wave * 16;

  int arow = m0 + (lane & 15); if (arow >= SEQ) arow = SEQ - 1;
  const int kb8 = (lane & 16) >> 1;        // A-frag K sub-offset (0 / 8)
  const unsigned short* xb = hbf + (size_t)b*SEQ*DM + (size_t)h*HDm;

  v16bf af[2];
  #pragma unroll
  for (int kk = 0; kk < 2; ++kk){
    BFV r;
    const unsigned short* p = xb + (size_t)arow*DM + kk*32 + kb8;
    r.h[0] = *(const v8us*)(p);
    r.h[1] = *(const v8us*)(p + 16);
    af[kk] = r.b;
  }

  const size_t head = (size_t)bh * SEQ * HDm;
  qkv_one(af, wq, qo + head, m0, lane, false);
  qkv_one(af, wk, ko + head, m0, lane, false);
  qkv_one(af, wv, vo + head, m0, lane, true);   // transposed
}

// ---------------------------------------------------------------------------
// flash attention: block = 8 waves x 16 query rows, all same (b,h).
// K/V tiles (32 keys) double-buffered in LDS via GLOBAL_LOAD_ASYNC_TO_LDS,
// shared by all waves; one barrier per key step.
// Q,K bf16 [B,H,S,64]; Vt bf16 [B,H,64,S]; output bf16 [B,S,512] at col h*64.
// ---------------------------------------------------------------------------
__global__ __launch_bounds__(256)
void attn_kernel(const unsigned short* __restrict__ Q,
                 const unsigned short* __restrict__ K,
                 const unsigned short* __restrict__ Vt,
                 unsigned short* __restrict__ O){
  __shared__ __align__(16) unsigned short Kl[2][32*72];    // [key][e], pad 72
  __shared__ __align__(16) unsigned short Vl[2][64*40];    // [e][key], pad 40
  __shared__ __align__(16) unsigned short Pl[8][16*40];    // per-wave P tile
  const int tid  = threadIdx.x;
  const int lane = tid & 31;
  const int wave = tid >> 5;
  const int bh   = blockIdx.y;
  const int b    = bh >> 3, h = bh & 7;
  const int m0   = blockIdx.x * 128 + wave * 16;

  const unsigned short* qp  = Q  + (size_t)bh*SEQ*HDm;
  const unsigned short* kp  = K  + (size_t)bh*SEQ*HDm;
  const unsigned short* vtp = Vt + (size_t)bh*HDm*SEQ;

  int arow = m0 + (lane & 15); if (arow >= SEQ) arow = SEQ - 1;
  const int kb8 = (lane & 16) >> 1;

  v16bf qa[2];
  #pragma unroll
  for (int kk = 0; kk < 2; ++kk){
    BFV r;
    const unsigned short* p = qp + (size_t)arow*HDm + kk*32 + kb8;
    r.h[0] = *(const v8us*)(p);
    r.h[1] = *(const v8us*)(p + 16);
    qa[kk] = r.b;
  }

  // staging assignments (256 threads, 16B each per array)
  const int skey = tid >> 3;        // 0..31  K row
  const int se   = (tid & 7) * 8;   // 0..56  K col chunk
  const int ve   = tid >> 2;        // 0..63  V row (e-dim)
  const int vk   = (tid & 3) * 8;   // 0..24  V key chunk

  // issue async stage of tile kt0 into buffer `bf`
  auto stage = [&](int kt0, int bf){
    int kg = kt0 + skey; if (kg >= SEQ) kg = SEQ - 1;          // finite rows
    async_copy_b128(kp + (size_t)kg*HDm + se, &Kl[bf][skey*72 + se]);
    int ks = kt0 + vk; if (ks + 8 > SEQ) ks = SEQ - 8;         // SEQ%8==0
    async_copy_b128(vtp + (size_t)ve*SEQ + ks, &Vl[bf][ve*40 + vk]);
  };

  float m8[8], l8[8];
  v8f oacc[4];
  #pragma unroll
  for (int e = 0; e < 8; ++e){ m8[e] = -3.0e38f; l8[e] = 0.0f; }
  #pragma unroll
  for (int j = 0; j < 4; ++j) oacc[j] = v8f{};

  unsigned short* pl = &Pl[wave][0];

  stage(0, 0);
  int buf = 0;
  for (int kt0 = 0; kt0 < SEQ; kt0 += 32){
    asm volatile("s_wait_asynccnt 0x0" ::: "memory");  // this buffer landed
    __syncthreads();                                    // all waves: landed +
                                                        // prev buffer free
    if (kt0 + 32 < SEQ) stage(kt0 + 32, buf ^ 1);       // prefetch next tile

    // ---- scores = Q Kt^T  (two 16x16 C tiles over the 32 keys) ----
    v8f sc[2]; sc[0] = v8f{}; sc[1] = v8f{};
    #pragma unroll
    for (int t = 0; t < 2; ++t){
      const unsigned short* kr = &Kl[buf][(t*16 + (lane & 15))*72 + kb8];
      #pragma unroll
      for (int kk = 0; kk < 2; ++kk){
        BFV r;
        r.h[0] = *(const v8us*)(kr + kk*32);
        r.h[1] = *(const v8us*)(kr + kk*32 + 16);
        sc[t] = wmma_bf16(qa[kk], r.b, sc[t]);
      }
    }

    // ---- online softmax update ----
    const int c0 = kt0 + (lane & 15);
    const int c1 = c0 + 16;
    float p0[8], p1[8];
    #pragma unroll
    for (int e = 0; e < 8; ++e){
      float s0 = (c0 < SEQ) ? sc[0][e] * 0.125f : -3.0e38f;   // 1/sqrt(64)
      float s1 = (c1 < SEQ) ? sc[1][e] * 0.125f : -3.0e38f;
      float mx = fmaxf(s0, s1);
      #pragma unroll
      for (int off = 1; off < 16; off <<= 1) mx = fmaxf(mx, __shfl_xor(mx, off));
      const float mnew  = fmaxf(m8[e], mx);
      const float scale = __expf(m8[e] - mnew);
      const float e0 = __expf(s0 - mnew);
      const float e1 = __expf(s1 - mnew);
      float rs = e0 + e1;
      #pragma unroll
      for (int off = 1; off < 16; off <<= 1) rs += __shfl_xor(rs, off);
      l8[e] = l8[e]*scale + rs;
      m8[e] = mnew;
      #pragma unroll
      for (int j = 0; j < 4; ++j) oacc[j][e] *= scale;
      p0[e] = e0; p1[e] = e1;
    }

    // ---- P: C layout -> LDS -> A layout ----
    {
      const int rb = (lane & 16) ? 8 : 0;
      const int cc = lane & 15;
      #pragma unroll
      for (int e = 0; e < 8; ++e){
        pl[(rb + e)*40 + cc]      = f2bf(p0[e]);
        pl[(rb + e)*40 + cc + 16] = f2bf(p1[e]);
      }
    }
    __asm__ volatile("s_wait_dscnt 0x0" ::: "memory");
    v16bf pf;
    {
      const int row = lane & 15;
      BFV r;
      #pragma unroll
      for (int j = 0; j < 8; ++j){
        r.u[j]     = pl[row*40 + kb8 + j];
        r.u[8 + j] = pl[row*40 + kb8 + 16 + j];
      }
      pf = r.b;
    }

    // ---- O += P x V  (V fragments from LDS, k 8h.. and 16+8h..) ----
    #pragma unroll
    for (int j = 0; j < 4; ++j){
      BFV r;
      const unsigned short* p = &Vl[buf][(j*16 + (lane & 15))*40 + kb8];
      r.h[0] = *(const v8us*)(p);
      r.h[1] = *(const v8us*)(p + 16);
      oacc[j] = wmma_bf16(pf, r.b, oacc[j]);
    }
    buf ^= 1;
  }

  // ---- epilogue: O /= l, store bf16 into [B,S,512] ----
  const int rb = (lane & 16) ? 8 : 0;
  const int cl = lane & 15;
  if (m0 + 16 <= SEQ){
    #pragma unroll
    for (int e = 0; e < 8; ++e){
      const int s = m0 + rb + e;
      const float inv = 1.0f / l8[e];
      #pragma unroll
      for (int j = 0; j < 4; ++j)
        O[((size_t)b*SEQ + s)*DM + h*HDm + j*16 + cl] = f2bf(oacc[j][e] * inv);
    }
  } else {
    #pragma unroll
    for (int e = 0; e < 8; ++e){
      const int s = m0 + rb + e;
      if (s < SEQ){
        const float inv = 1.0f / l8[e];
        #pragma unroll
        for (int j = 0; j < 4; ++j)
          O[((size_t)b*SEQ + s)*DM + h*HDm + j*16 + cl] = f2bf(oacc[j][e] * inv);
      }
    }
  }
}

// ---------------------------------------------------------------------------
// generic GEMM: C[M,N] = A[M,K](bf16) x W[N,K](bf16)^T + bias, opt. ReLU
// block tile 128x128 (8 waves, each 32x64), K step 32. M,N,K mult. of 128/32.
// ---------------------------------------------------------------------------
__global__ __launch_bounds__(256)
void gemm_kernel(const unsigned short* __restrict__ A,
                 const unsigned short* __restrict__ W,
                 const float* __restrict__ bias,
                 float* __restrict__ outF,
                 unsigned short* __restrict__ outB,
                 int M, int N, int K, int relu){
  const int lane = threadIdx.x & 31;
  const int wave = threadIdx.x >> 5;
  const int wm = wave & 3, wn = wave >> 2;
  const int m0 = blockIdx.y*128 + wm*32;
  const int n0 = blockIdx.x*128 + wn*64;

  v8f acc[2][4];
  #pragma unroll
  for (int i = 0; i < 2; ++i)
    #pragma unroll
    for (int j = 0; j < 4; ++j) acc[i][j] = v8f{};

  const int arow = lane & 15;
  const int kb8  = (lane & 16) >> 1;
  const int kb16 = lane & 16;
  const int cl   = lane & 15;

  for (int k0 = 0; k0 < K; k0 += 32){
    v16bf a0, a1;
    {
      BFV r;
      const unsigned short* p = A + (size_t)(m0 + arow)*K + k0 + kb8;
      r.h[0] = *(const v8us*)(p); r.h[1] = *(const v8us*)(p + 16);
      a0 = r.b;
      const unsigned short* p2 = A + (size_t)(m0 + 16 + arow)*K + k0 + kb8;
      r.h[0] = *(const v8us*)(p2); r.h[1] = *(const v8us*)(p2 + 16);
      a1 = r.b;
    }
    #pragma unroll
    for (int j = 0; j < 4; ++j){
      BFV r;
      const unsigned short* p = W + (size_t)(n0 + j*16 + cl)*K + k0 + kb8;
      r.h[0] = *(const v8us*)(p); r.h[1] = *(const v8us*)(p + 16);
      acc[0][j] = wmma_bf16(a0, r.b, acc[0][j]);
      acc[1][j] = wmma_bf16(a1, r.b, acc[1][j]);
    }
    if (k0 + 32 < K){   // gfx1250 global_prefetch_b8 of next k-tile
      __builtin_prefetch(A + (size_t)(m0 + arow)*K + k0 + 32 + kb8, 0, 1);
      __builtin_prefetch(W + (size_t)(n0 + cl)*K + k0 + 32 + kb16, 0, 1);
    }
  }

  const int rb = (lane & 16) ? 8 : 0;
  #pragma unroll
  for (int i = 0; i < 2; ++i){
    #pragma unroll
    for (int j = 0; j < 4; ++j){
      const int col = n0 + j*16 + cl;
      const float bv = bias ? bias[col] : 0.0f;
      #pragma unroll
      for (int e = 0; e < 8; ++e){
        const int row = m0 + i*16 + rb + e;
        float v = acc[i][j][e] + bv;
        if (relu) v = fmaxf(v, 0.0f);
        const size_t idx = (size_t)row*N + col;
        if (outF) outF[idx] = v;
        if (outB) outB[idx] = f2bf(v);
      }
    }
  }
}

// ---------------------------------------------------------------------------
// y = LayerNorm(a + b) * g + be ; one wave per 512-element row
// ---------------------------------------------------------------------------
__global__ __launch_bounds__(256)
void add_ln_kernel(const float* __restrict__ a, const float* __restrict__ b,
                   const float* __restrict__ g, const float* __restrict__ be,
                   float* __restrict__ outF, unsigned short* __restrict__ outB){
  const int row  = blockIdx.x*8 + (threadIdx.x >> 5);
  const int lane = threadIdx.x & 31;
  const size_t base = (size_t)row * DM;
  float x[16];
  float s = 0.0f;
  #pragma unroll
  for (int i = 0; i < 16; ++i){
    x[i] = a[base + lane + i*32] + b[base + lane + i*32];
    s += x[i];
  }
  #pragma unroll
  for (int off = 16; off > 0; off >>= 1) s += __shfl_xor(s, off);
  const float mean = s * (1.0f/DM);
  float vs = 0.0f;
  #pragma unroll
  for (int i = 0; i < 16; ++i){ float d = x[i] - mean; vs += d*d; }
  #pragma unroll
  for (int off = 16; off > 0; off >>= 1) vs += __shfl_xor(vs, off);
  const float r = rsqrtf(vs*(1.0f/DM) + 1e-5f);
  #pragma unroll
  for (int i = 0; i < 16; ++i){
    const int c = lane + i*32;
    const float y = (x[i] - mean)*r*g[c] + be[c];
    outF[base + c] = y;
    outB[base + c] = f2bf(y);
  }
}

// ---------------------------------------------------------------------------
extern "C" void kernel_launch(void* const* d_in, const int* in_sizes, int n_in,
                              void* d_out, int out_size, void* d_ws, size_t ws_size,
                              hipStream_t stream){
  (void)in_sizes; (void)n_in; (void)out_size; (void)ws_size;
  const int*   x    = (const int*)  d_in[0];
  const float* emb  = (const float*)d_in[1];
  const float* pe   = (const float*)d_in[2];
  const float* Wq   = (const float*)d_in[3];
  const float* Wk   = (const float*)d_in[4];
  const float* Wv   = (const float*)d_in[5];
  const float* Wo   = (const float*)d_in[6];
  const float* bo   = (const float*)d_in[7];
  const float* ln1w = (const float*)d_in[8];
  const float* ln1b = (const float*)d_in[9];
  const float* W1   = (const float*)d_in[10];
  const float* b1   = (const float*)d_in[11];
  const float* W2   = (const float*)d_in[12];
  const float* b2   = (const float*)d_in[13];
  const float* ln2w = (const float*)d_in[14];
  const float* ln2b = (const float*)d_in[15];

  char* ws = (char*)d_ws;
  size_t off = 0;
  auto alloc = [&](size_t bytes)->char*{
    char* p = ws + off; off = (off + bytes + 255) & ~(size_t)255; return p;
  };
  unsigned short* wqb  = (unsigned short*)alloc((size_t)NL*HDm*HDm*2);
  unsigned short* wkb  = (unsigned short*)alloc((size_t)NL*HDm*HDm*2);
  unsigned short* wvb  = (unsigned short*)alloc((size_t)NL*HDm*HDm*2);
  unsigned short* wob  = (unsigned short*)alloc((size_t)NL*DM*DM*2);
  unsigned short* w1b  = (unsigned short*)alloc((size_t)NL*EXPF*DM*DM*2);
  unsigned short* w2b  = (unsigned short*)alloc((size_t)NL*EXPF*DM*DM*2);
  float*          hf   = (float*)         alloc((size_t)MROWS*DM*4);
  unsigned short* hbfp = (unsigned short*)alloc((size_t)MROWS*DM*2);
  unsigned short* qb   = (unsigned short*)alloc((size_t)MROWS*DM*2);
  unsigned short* kb   = (unsigned short*)alloc((size_t)MROWS*DM*2);
  unsigned short* vb   = (unsigned short*)alloc((size_t)MROWS*DM*2);  // Vt [B,H,64,S]
  unsigned short* abf  = (unsigned short*)alloc((size_t)MROWS*DM*2);
  float*          t0   = (float*)         alloc((size_t)MROWS*DM*4);
  unsigned short* mid  = (unsigned short*)alloc((size_t)MROWS*DM*EXPF*2);

  auto conv = [&](const float* in, unsigned short* outp, long n){
    cvt_kernel<<<dim3((unsigned)((n + 255)/256)), dim3(256), 0, stream>>>(in, outp, n);
  };
  conv(Wq, wqb, (long)NL*HDm*HDm);
  conv(Wk, wkb, (long)NL*HDm*HDm);
  conv(Wv, wvb, (long)NL*HDm*HDm);
  conv(Wo, wob, (long)NL*DM*DM);
  conv(W1, w1b, (long)NL*EXPF*DM*DM);
  conv(W2, w2b, (long)NL*EXPF*DM*DM);

  embed_kernel<<<dim3(MROWS), dim3(256), 0, stream>>>(x, emb, pe, hf, hbfp);

  for (int l = 0; l < NL; ++l){
    const unsigned short* wq_l = wqb + (size_t)l*HDm*HDm;
    const unsigned short* wk_l = wkb + (size_t)l*HDm*HDm;
    const unsigned short* wv_l = wvb + (size_t)l*HDm*HDm;
    const unsigned short* wo_l = wob + (size_t)l*DM*DM;
    const unsigned short* w1_l = w1b + (size_t)l*EXPF*DM*DM;
    const unsigned short* w2_l = w2b + (size_t)l*EXPF*DM*DM;

    qkv_kernel<<<dim3(6, Bn*NH), dim3(256), 0, stream>>>(
        hbfp, wq_l, wk_l, wv_l, qb, kb, vb);

    attn_kernel<<<dim3(6, Bn*NH), dim3(256), 0, stream>>>(qb, kb, vb, abf);

    // o @ Wo^T + bo -> t0 (f32)
    gemm_kernel<<<dim3(DM/128, MROWS/128), dim3(256), 0, stream>>>(
        abf, wo_l, bo + (size_t)l*DM, t0, (unsigned short*)nullptr,
        MROWS, DM, DM, 0);

    // n1 = LN(t0 + h) -> hf (f32) + abf (bf16)
    add_ln_kernel<<<dim3(MROWS/8), dim3(256), 0, stream>>>(
        t0, hf, ln1w + (size_t)l*DM, ln1b + (size_t)l*DM, hf, abf);

    // relu(n1 @ W1^T + b1) -> mid (bf16)
    gemm_kernel<<<dim3(EXPF*DM/128, MROWS/128), dim3(256), 0, stream>>>(
        abf, w1_l, b1 + (size_t)l*EXPF*DM, (float*)nullptr, mid,
        MROWS, EXPF*DM, DM, 1);

    // mid @ W2^T + b2 -> t0 (f32)
    gemm_kernel<<<dim3(DM/128, MROWS/128), dim3(256), 0, stream>>>(
        mid, w2_l, b2 + (size_t)l*DM, t0, (unsigned short*)nullptr,
        MROWS, DM, EXPF*DM, 0);

    // h = LN(t0 + n1) -> hf/d_out (f32) + hbf (bf16)
    float* outF = (l == NL-1) ? (float*)d_out : hf;
    add_ln_kernel<<<dim3(MROWS/8), dim3(256), 0, stream>>>(
        t0, hf, ln2w + (size_t)l*DM, ln2b + (size_t)l*DM, outF, hbfp);
  }
}
